// SKITNO2d_19447611916892
// MI455X (gfx1250) — matrix operand order repaired and checked by reading, not compiled, not run on hardware
//
#include <hip/hip_runtime.h>
#include <hip/hip_bf16.h>
#include <math.h>

// ---------------------------------------------------------------------------
// SKI-Toeplitz Neural Operator 2D — CDNA5 (gfx1250) implementation.
// All matmuls run on v_wmma_f32_16x16x32_bf16 (wave32 WMMA, f32 accum).
// Big GEMMs use double-buffered LDS staging (block-level A/B reuse).
// ---------------------------------------------------------------------------

typedef __attribute__((ext_vector_type(16))) __bf16 bf16x16;
typedef __attribute__((ext_vector_type(8)))  __bf16 bf16x8;
typedef __attribute__((ext_vector_type(8)))  float  f32x8;
typedef __attribute__((ext_vector_type(4)))  float  f32x4;

#define DEVFN __device__ __forceinline__

static constexpr int BATCH  = 8;
static constexpr int NSEQ   = 4096;   // H*W
static constexpr int DMODEL = 512;
static constexpr int G      = 1024;   // D1 = heads * head_dim
static constexpr int HEADS  = 8;
static constexpr int HD     = 128;
static constexpr int NK     = 16;
static constexpr int RANK   = 32;
static constexpr int MROWS  = BATCH * NSEQ;  // 32768

// ---------------- WMMA fragment helpers (per CDNA5 ISA 7.12.2) -------------

// A-matrix 16x32 bf16: lane L holds row m=L%16; kbase=(L/16)*8;
// VGPR0..3 -> K=kbase..kbase+7, VGPR4..7 -> K=kbase+16..kbase+23.
DEVFN bf16x16 load_a_frag(const __bf16* base, int ld) {
  int lane = threadIdx.x & 31;
  const __bf16* p = base + (size_t)(lane & 15) * ld + ((lane >> 4) << 3);
  bf16x8 lo = *(const bf16x8*)(p);
  bf16x8 hi = *(const bf16x8*)(p + 16);
  bf16x16 f;
#pragma unroll
  for (int i = 0; i < 8; ++i) { f[i] = lo[i]; f[i + 8] = hi[i]; }
  return f;
}

// B-matrix 32x16 bf16: lane L holds column n=L%16; kbase=(L/16)*8;
// VGPR0..3 -> K=kbase..kbase+7, VGPR4..7 -> K=kbase+16..kbase+23.
// Source layout is row-major Bsrc[n][k] (i.e. B[k][n] transposed).
DEVFN bf16x16 load_b_frag(const __bf16* base, int ld) {
  int lane = threadIdx.x & 31;
  const __bf16* p = base + (size_t)(lane & 15) * ld + ((lane >> 4) << 3);
  bf16x8 lo = *(const bf16x8*)(p);
  bf16x8 hi = *(const bf16x8*)(p + 16);
  bf16x16 f;
#pragma unroll
  for (int i = 0; i < 8; ++i) { f[i] = lo[i]; f[i + 8] = hi[i]; }
  return f;
}

DEVFN f32x8 wmma_bf16(bf16x16 a, bf16x16 b, f32x8 c) {
  return __builtin_amdgcn_wmma_f32_16x16x32_bf16(
      /*neg_a=*/false, a, /*neg_b=*/false, b,
      /*c_mod=*/(short)0, c, /*reuse_a=*/false, /*reuse_b=*/false);
}

DEVFN f32x8 zero8() { f32x8 z = {0.f,0.f,0.f,0.f,0.f,0.f,0.f,0.f}; return z; }

DEVFN float silu_f(float v) { return v * (1.f / (1.f + __expf(-v))); }

// ---------------- elementwise casts ----------------------------------------

__global__ void __launch_bounds__(256)
cast_f32_bf16(const float* __restrict__ in, __bf16* __restrict__ out, int n) {
  int i = blockIdx.x * 256 + threadIdx.x;
  if (i < n) out[i] = (__bf16)in[i];
}

// ---------------- Toeplitz kernel construction -----------------------------

// a[h][l][d], l in [0,128): l=0 zero-lag, l in [1,64) positive lags (decayed,
// interpolated from NK inducing points), l=64 zero, l in (64,128) negative
// lag t(l-128).
__global__ void __launch_bounds__(256)
build_lags(const float* __restrict__ zlag,
           const float* __restrict__ pa, const float* __restrict__ pb,
           const float* __restrict__ na, const float* __restrict__ nb,
           float* __restrict__ a) {
  int tid = blockIdx.x * 256 + threadIdx.x;
  if (tid >= HEADS * 128 * HD) return;
  int h = tid / (128 * HD);
  int l = (tid / HD) & 127;
  int d = tid % HD;
  float val;
  if (l == 0) {
    val = zlag[h * HD + d];
  } else if (l == 64) {
    val = 0.f;
  } else {
    int j;
    const float* Am; const float* Bm;
    if (l < 64) { j = l;        Am = pa; Bm = pb; }
    else        { j = 128 - l;  Am = na; Bm = nb; }
    float pos = (float)(j - 1) * ((float)(NK - 1) / 62.0f);
    int lo = (int)floorf(pos);
    int hi = lo + 1; if (hi > NK - 1) hi = NK - 1;
    float w = pos - (float)lo;
    float tlo = 0.f, thi = 0.f;
#pragma unroll 8
    for (int r = 0; r < RANK; ++r) {
      float bv = Bm[((size_t)h * RANK + r) * HD + d];
      tlo += Am[((size_t)h * NK + lo) * RANK + r] * bv;
      thi += Am[((size_t)h * NK + hi) * RANK + r] * bv;
    }
    float decay = __expf((float)j * -0.0010005003335835335f);  // ln(0.999)
    val = ((1.f - w) * tlo + w * thi) * decay;
  }
  a[tid] = val;
}

// Tmat[c][n][k] = bf16( a[h][(n-k)&127][d] ), c = h*HD + d.  (row n = output
// index, contiguous k = reduction index -> ready for load_a/load_b frags)
__global__ void __launch_bounds__(256)
build_T(const float* __restrict__ a, __bf16* __restrict__ T) {
  int tid = blockIdx.x * 256 + threadIdx.x;
  if (tid >= G * 4096) return;
  int c = tid >> 12;
  int n = (tid >> 6) & 63;
  int k = tid & 63;
  int h = c >> 7;
  int d = c & 127;
  int l = (n - k) & 127;
  T[tid] = (__bf16)a[((size_t)h * 128 + l) * HD + d];
}

// ---------------- WMMA GEMM: out = act(A @ W^T + bias) ---------------------
// A: bf16 row-major [M][K]; W: bf16 row-major [N][K]; bias: f32 [N].
// Block tile 128x64 (8 waves, each 32x32), double-buffered LDS staging.
// out_mode 0: f32 natural row-major [M][N]
// out_mode 1: bf16 batch-transposed out[b][n][hw], b=m/4096, hw=m%4096
__global__ void __launch_bounds__(256)
gemm_bias_act(const __bf16* __restrict__ A, const __bf16* __restrict__ W,
              const float* __restrict__ bias, void* __restrict__ out,
              int N, int K, int act_silu, int out_mode) {
  constexpr int PITCH = 40;  // 32 bf16 + 8 pad -> staggers LDS banks
  __shared__ alignas(16) __bf16 ldsA[2][128 * PITCH];
  __shared__ alignas(16) __bf16 ldsB[2][64 * PITCH];

  int tid  = threadIdx.x;
  int wave = tid >> 5;
  int lane = tid & 31;
  int bm = blockIdx.y * 128;
  int bn = blockIdx.x * 64;
  int wm = (wave >> 1) * 32;   // wave tile inside block
  int wn = (wave & 1) * 32;

  // Cooperative staging map: each thread owns one 16B chunk per row group.
  int lrow = tid >> 2;            // 0..63
  int lcol = (tid & 3) * 8;       // bf16 column offset within 32-wide tile
  const __bf16* gA0 = A + (size_t)(bm + lrow) * K + lcol;
  const __bf16* gA1 = A + (size_t)(bm + lrow + 64) * K + lcol;
  const __bf16* gB  = W + (size_t)(bn + lrow) * K + lcol;

  // Preload k-step 0 and stage into buffer 0.
  bf16x8 ra0 = *(const bf16x8*)(gA0);
  bf16x8 ra1 = *(const bf16x8*)(gA1);
  bf16x8 rb  = *(const bf16x8*)(gB);
  *(bf16x8*)&ldsA[0][(size_t)lrow * PITCH + lcol]        = ra0;
  *(bf16x8*)&ldsA[0][(size_t)(lrow + 64) * PITCH + lcol] = ra1;
  *(bf16x8*)&ldsB[0][(size_t)lrow * PITCH + lcol]        = rb;
  __syncthreads();

  f32x8 acc[2][2];
  acc[0][0] = zero8(); acc[0][1] = zero8();
  acc[1][0] = zero8(); acc[1][1] = zero8();

  int nsteps = K >> 5;
  int buf = 0;
  for (int s = 0; s < nsteps; ++s) {
    // Issue next tile's global loads early (hidden under WMMA work).
    int kn = (s + 1) << 5;
    if (s + 1 < nsteps) {
      ra0 = *(const bf16x8*)(gA0 + kn);
      ra1 = *(const bf16x8*)(gA1 + kn);
      rb  = *(const bf16x8*)(gB + kn);
      if (s + 2 < nsteps) {
        __builtin_prefetch(gA0 + kn + 32, 0, 1);   // global_prefetch_b8
        __builtin_prefetch(gB + kn + 32, 0, 1);
      }
    }
    // WMMA fragments from LDS (ds_load_b128).
    const __bf16* la = &ldsA[buf][0];
    const __bf16* lb = &ldsB[buf][0];
    bf16x16 a0 = load_a_frag(la + (size_t)wm * PITCH, PITCH);
    bf16x16 a1 = load_a_frag(la + (size_t)(wm + 16) * PITCH, PITCH);
    bf16x16 b0 = load_b_frag(lb + (size_t)wn * PITCH, PITCH);
    bf16x16 b1 = load_b_frag(lb + (size_t)(wn + 16) * PITCH, PITCH);
    acc[0][0] = wmma_bf16(a0, b0, acc[0][0]);
    acc[0][1] = wmma_bf16(a0, b1, acc[0][1]);
    acc[1][0] = wmma_bf16(a1, b0, acc[1][0]);
    acc[1][1] = wmma_bf16(a1, b1, acc[1][1]);
    // Stage next tile into the other buffer; one barrier per step.
    if (s + 1 < nsteps) {
      int nb = buf ^ 1;
      *(bf16x8*)&ldsA[nb][(size_t)lrow * PITCH + lcol]        = ra0;
      *(bf16x8*)&ldsA[nb][(size_t)(lrow + 64) * PITCH + lcol] = ra1;
      *(bf16x8*)&ldsB[nb][(size_t)lrow * PITCH + lcol]        = rb;
      __syncthreads();
      buf = nb;
    }
  }

  int gm = bm + wm;
  int gn = bn + wn;
#pragma unroll
  for (int mt = 0; mt < 2; ++mt)
#pragma unroll
    for (int nt = 0; nt < 2; ++nt) {
      int n     = gn + nt * 16 + (lane & 15);
      int mbase = gm + mt * 16 + ((lane >> 4) << 3);
      float bv = bias[n];
      float vals[8];
#pragma unroll
      for (int r = 0; r < 8; ++r) {
        float v = acc[mt][nt][r] + bv;
        vals[r] = act_silu ? silu_f(v) : v;
      }
      if (out_mode == 0) {
        float* po = (float*)out;
#pragma unroll
        for (int r = 0; r < 8; ++r)
          po[(size_t)(mbase + r) * N + n] = vals[r];
      } else {
        __bf16* po = (__bf16*)out + (size_t)(mbase >> 12) * ((size_t)N * 4096)
                     + (size_t)n * 4096 + (mbase & 4095);
        bf16x8 v;
#pragma unroll
        for (int r = 0; r < 8; ++r) v[r] = (__bf16)vals[r];
        *(bf16x8*)po = v;
      }
    }
}

// ---------------- batched 64x64 per-channel Toeplitz matmul ----------------
// For each bc = b*G + c: D = X(64x64, rows=m over contiguous k) @ T_c^T.
// store_mode 0: bf16 transposed  out[bc*4096 + n*64 + m]
// store_mode 1: f32  transposed  out[bc*4096 + n*64 + m]
// store_mode 2: f32  natural acc out[bc*4096 + m*64 + n] += v
__global__ void __launch_bounds__(256)
mix64(const __bf16* __restrict__ X, const __bf16* __restrict__ T,
      void* __restrict__ out, int store_mode) {
  int lane = threadIdx.x & 31;
  int gw = blockIdx.x * 8 + (threadIdx.x >> 5);
  int bc  = gw >> 2;
  int sub = gw & 3;
  int wm = (sub >> 1) * 32;
  int wn = (sub & 1) * 32;
  int c = bc & (G - 1);

  const __bf16* Xb = X + (size_t)bc * 4096;
  const __bf16* Tb = T + (size_t)c * 4096;

  f32x8 acc[2][2];
  acc[0][0] = zero8(); acc[0][1] = zero8();
  acc[1][0] = zero8(); acc[1][1] = zero8();

#pragma unroll
  for (int k = 0; k < 64; k += 32) {
    bf16x16 a0 = load_a_frag(Xb + (size_t)wm * 64 + k, 64);
    bf16x16 a1 = load_a_frag(Xb + (size_t)(wm + 16) * 64 + k, 64);
    bf16x16 b0 = load_b_frag(Tb + (size_t)wn * 64 + k, 64);
    bf16x16 b1 = load_b_frag(Tb + (size_t)(wn + 16) * 64 + k, 64);
    acc[0][0] = wmma_bf16(a0, b0, acc[0][0]);
    acc[0][1] = wmma_bf16(a0, b1, acc[0][1]);
    acc[1][0] = wmma_bf16(a1, b0, acc[1][0]);
    acc[1][1] = wmma_bf16(a1, b1, acc[1][1]);
  }

#pragma unroll
  for (int mt = 0; mt < 2; ++mt)
#pragma unroll
    for (int nt = 0; nt < 2; ++nt) {
      int n     = wn + nt * 16 + (lane & 15);
      int mbase = wm + mt * 16 + ((lane >> 4) << 3);
      if (store_mode == 0) {
        __bf16* po = (__bf16*)out + (size_t)bc * 4096 + (size_t)n * 64 + mbase;
        bf16x8 v;
#pragma unroll
        for (int r = 0; r < 8; ++r) v[r] = (__bf16)acc[mt][nt][r];
        *(bf16x8*)po = v;
      } else if (store_mode == 1) {
        float* po = (float*)out + (size_t)bc * 4096 + (size_t)n * 64 + mbase;
        f32x4 v0 = {acc[mt][nt][0], acc[mt][nt][1], acc[mt][nt][2], acc[mt][nt][3]};
        f32x4 v1 = {acc[mt][nt][4], acc[mt][nt][5], acc[mt][nt][6], acc[mt][nt][7]};
        *(f32x4*)po = v0;
        *(f32x4*)(po + 4) = v1;
      } else {
        float* po = (float*)out + (size_t)bc * 4096;
#pragma unroll
        for (int r = 0; r < 8; ++r)
          po[(size_t)(mbase + r) * 64 + n] += acc[mt][nt][r];
      }
    }
}

// ---------------- batched 64x64 bf16 transpose via LDS ---------------------
__global__ void __launch_bounds__(256)
transpose64(const __bf16* __restrict__ in, __bf16* __restrict__ out) {
  __shared__ __bf16 lds[64 * 65];
  size_t base = (size_t)blockIdx.x * 4096;
#pragma unroll
  for (int j = 0; j < 16; ++j) {
    int idx = threadIdx.x + j * 256;
    lds[(idx & 63) * 65 + (idx >> 6)] = in[base + idx];
  }
  __syncthreads();
#pragma unroll
  for (int j = 0; j < 16; ++j) {
    int idx = threadIdx.x + j * 256;
    out[base + idx] = lds[(idx >> 6) * 65 + (idx & 63)];
  }
}

// ---------------- gate: gi[b][hw][c] = ut[b][c][hw] * o[b][c][hw] ----------
// Fused multiply + LDS tile transpose back to token-major bf16.
__global__ void __launch_bounds__(256)
gate_transpose(const __bf16* __restrict__ ut, const float* __restrict__ o,
               __bf16* __restrict__ gi) {
  __shared__ __bf16 lds[64 * 65];
  int bid = blockIdx.x;
  int b  = bid >> 10;          // 1024 tiles per batch (16 c-tiles x 64 hw-tiles)
  int ct = (bid >> 6) & 15;
  int nt = bid & 63;
  int c0 = ct * 64, hw0 = nt * 64;
  size_t cb = (size_t)b * G * NSEQ;
#pragma unroll
  for (int j = 0; j < 16; ++j) {
    int idx = threadIdx.x + j * 256;
    int cr = idx >> 6, hwc = idx & 63;
    size_t src = cb + (size_t)(c0 + cr) * NSEQ + hw0 + hwc;
    float v = (float)ut[src] * o[src];
    lds[hwc * 65 + cr] = (__bf16)v;
  }
  __syncthreads();
#pragma unroll
  for (int j = 0; j < 16; ++j) {
    int idx = threadIdx.x + j * 256;
    int hwr = idx >> 6, cc = idx & 63;
    gi[(size_t)b * NSEQ * G + (size_t)(hw0 + hwr) * G + c0 + cc] =
        lds[hwr * 65 + cc];
  }
}

// ---------------------------------------------------------------------------

extern "C" void kernel_launch(void* const* d_in, const int* in_sizes, int n_in,
                              void* d_out, int out_size, void* d_ws, size_t ws_size,
                              hipStream_t stream) {
  (void)in_sizes; (void)n_in; (void)out_size; (void)ws_size;
  const float* x    = (const float*)d_in[0];
  const float* u_w  = (const float*)d_in[1];
  const float* u_b  = (const float*)d_in[2];
  const float* v_w  = (const float*)d_in[3];
  const float* v_b  = (const float*)d_in[4];
  const float* o_w  = (const float*)d_in[5];
  const float* o_b  = (const float*)d_in[6];
  const float* t1z  = (const float*)d_in[7];
  const float* t1pa = (const float*)d_in[8];
  const float* t1pb = (const float*)d_in[9];
  const float* t1na = (const float*)d_in[10];
  const float* t1nb = (const float*)d_in[11];
  const float* t2z  = (const float*)d_in[12];
  const float* t2pa = (const float*)d_in[13];
  const float* t2pb = (const float*)d_in[14];
  const float* t2na = (const float*)d_in[15];
  const float* t2nb = (const float*)d_in[16];

  char* wp = (char*)d_ws;
  auto alloc = [&](size_t bytes) -> char* {
    char* p = wp;
    wp += (bytes + 255) & ~(size_t)255;
    return p;
  };
  __bf16* x_bf  = (__bf16*)alloc((size_t)MROWS * DMODEL * 2);
  __bf16* uw_bf = (__bf16*)alloc((size_t)G * DMODEL * 2);
  __bf16* vw_bf = (__bf16*)alloc((size_t)G * DMODEL * 2);
  __bf16* ow_bf = (__bf16*)alloc((size_t)DMODEL * G * 2);
  float*  a1    = (float*) alloc((size_t)HEADS * 128 * HD * 4);
  float*  a2    = (float*) alloc((size_t)HEADS * 128 * HD * 4);
  __bf16* T1    = (__bf16*)alloc((size_t)G * 4096 * 2);
  __bf16* T2    = (__bf16*)alloc((size_t)G * 4096 * 2);
  __bf16* ut    = (__bf16*)alloc((size_t)MROWS * G * 2);
  __bf16* vt    = (__bf16*)alloc((size_t)MROWS * G * 2);
  __bf16* vtT   = (__bf16*)alloc((size_t)MROWS * G * 2);
  __bf16* y1t   = (__bf16*)alloc((size_t)MROWS * G * 2);
  __bf16* y2    = (__bf16*)alloc((size_t)MROWS * G * 2);
  float*  o12   = (float*) alloc((size_t)MROWS * G * 4);
  __bf16* gi    = (__bf16*)alloc((size_t)MROWS * G * 2);

  // 1) casts to bf16
  cast_f32_bf16<<<(MROWS * DMODEL + 255) / 256, 256, 0, stream>>>(x, x_bf, MROWS * DMODEL);
  cast_f32_bf16<<<(G * DMODEL + 255) / 256, 256, 0, stream>>>(u_w, uw_bf, G * DMODEL);
  cast_f32_bf16<<<(G * DMODEL + 255) / 256, 256, 0, stream>>>(v_w, vw_bf, G * DMODEL);
  cast_f32_bf16<<<(DMODEL * G + 255) / 256, 256, 0, stream>>>(o_w, ow_bf, DMODEL * G);

  // 2) build Toeplitz lag tables and per-channel 64x64 matrices
  int nlag = HEADS * 128 * HD;
  build_lags<<<(nlag + 255) / 256, 256, 0, stream>>>(t1z, t1pa, t1pb, t1na, t1nb, a1);
  build_lags<<<(nlag + 255) / 256, 256, 0, stream>>>(t2z, t2pa, t2pb, t2na, t2nb, a2);
  build_T<<<(G * 4096 + 255) / 256, 256, 0, stream>>>(a1, T1);
  build_T<<<(G * 4096 + 255) / 256, 256, 0, stream>>>(a2, T2);

  // 3) u,v projections (WMMA GEMMs), stored channel-major: ut/vt[b][c][hw]
  dim3 gUV(G / 64, MROWS / 128);
  gemm_bias_act<<<gUV, 256, 0, stream>>>(x_bf, uw_bf, u_b, ut, G, DMODEL, 1, 1);
  gemm_bias_act<<<gUV, 256, 0, stream>>>(x_bf, vw_bf, v_b, vt, G, DMODEL, 1, 1);

  // 4) per-(b,c) 64x64 transpose for path 2
  transpose64<<<BATCH * G, 256, 0, stream>>>(vt, vtT);

  // 5) four Toeplitz mixings (batched WMMA 64x64 matmuls)
  int mixBlocks = (BATCH * G * 4) / 8;  // 4 waves per bc, 8 waves per block
  mix64<<<mixBlocks, 256, 0, stream>>>(vt,  T1, y1t, 0);  // W-mix -> y1t[w'][h]
  mix64<<<mixBlocks, 256, 0, stream>>>(y1t, T2, o12, 1);  // H-mix -> o[h'][w']
  mix64<<<mixBlocks, 256, 0, stream>>>(vtT, T2, y2,  0);  // H-mix -> y2[h'][w]
  mix64<<<mixBlocks, 256, 0, stream>>>(y2,  T1, o12, 2);  // W-mix += o[h'][w'']

  // 6) gate + transpose back to token-major
  gate_transpose<<<BATCH * 1024, 256, 0, stream>>>(ut, o12, gi);

  // 7) output projection (WMMA GEMM), f32 natural store to d_out
  dim3 gO(DMODEL / 64, MROWS / 128);
  gemm_bias_act<<<gO, 256, 0, stream>>>(gi, ow_bf, o_b, d_out, DMODEL, G, 0, 0);
}
